// LlamaAttention_73349451481406
// MI455X (gfx1250) — matrix operand order repaired and checked by reading, not compiled, not run on hardware
//
#include <hip/hip_runtime.h>
#include <hip/hip_bf16.h>

#define Bsz   2
#define Sq    2048
#define HIDN  2048
#define NH    32
#define Dh    64
#define ALEN  10
#define BSROWS (Bsz*Sq)          // 4096
#define SCALE 0.125f             // 1/sqrt(64)

typedef float v8f   __attribute__((ext_vector_type(8)));
typedef __bf16 v16bf __attribute__((ext_vector_type(16)));
typedef unsigned int u32x4 __attribute__((ext_vector_type(4)));

union Frag { u32x4 q[2]; v16bf v; };

__device__ __forceinline__ unsigned short f2bf(float f) {
    unsigned int x = __float_as_uint(f);
    unsigned int r = x + 0x7FFFu + ((x >> 16) & 1u);   // RNE
    return (unsigned short)(r >> 16);
}
__device__ __forceinline__ float bf2f(unsigned short u) {
    return __uint_as_float(((unsigned int)u) << 16);
}

__device__ __forceinline__ v8f wmma_bf16(v16bf a, v16bf b, v8f c) {
    return __builtin_amdgcn_wmma_f32_16x16x32_bf16(
        false, a, false, b, (short)0, c, false, false);
}

// A operand: M x K row-major bf16. Lane layout: m = lane&15,
// elems 0..7 -> K = (lane>>4)*8 + 0..7, elems 8..15 -> +16.
__device__ __forceinline__ v16bf load_frag_A(const unsigned short* base, int ld,
                                             int m0, int k0, int lane) {
    int row = m0 + (lane & 15);
    int kb  = (lane >> 4) * 8;
    const unsigned short* p = base + (size_t)row * ld + k0 + kb;
    Frag f;
    f.q[0] = *(const u32x4*)(p);
    f.q[1] = *(const u32x4*)(p + 16);
    return f.v;
}

// B operand supplied as Bt: N x K row-major bf16 (i.e. B transposed).
// Lane layout: n = lane&15, elems 0..7 -> K = (lane>>4)*8 + 0..7, elems 8..15 -> +16.
__device__ __forceinline__ v16bf load_frag_B(const unsigned short* bt, int ld,
                                             int n0, int k0, int lane) {
    int n  = n0 + (lane & 15);
    int kb = (lane >> 4) * 8;
    const unsigned short* p = bt + (size_t)n * ld + k0 + kb;
    Frag f;
    f.q[0] = *(const u32x4*)(p);
    f.q[1] = *(const u32x4*)(p + 16);
    return f.v;
}

// ---------------- elementwise conversion kernels ----------------

__global__ void cvt_bf16_kernel(const float* __restrict__ in,
                                unsigned short* __restrict__ out, int n) {
    int t = blockIdx.x * blockDim.x + threadIdx.x;
    if (t < n) out[t] = f2bf(in[t]);
}

// W [K,N] fp32 row-major -> Wt [N,K] bf16 row-major
__global__ void transpose_cvt_kernel(const float* __restrict__ W,
                                     unsigned short* __restrict__ Wt) {
    int t = blockIdx.x * blockDim.x + threadIdx.x;   // t = n*K + k
    int k = t & (HIDN - 1);
    int n = t >> 11;
    Wt[(size_t)n * HIDN + k] = f2bf(W[(size_t)k * HIDN + n]);
}

// ---------------- bf16 WMMA GEMM: C[M,N] = A[M,K] * Bt[N,K]^T ----------------
// grid: (N/64, M/32), block: 32 (one wave computes a 32x64 tile)
__global__ __launch_bounds__(32)
void gemm_bt_kernel(const unsigned short* __restrict__ A,
                    const unsigned short* __restrict__ Bt,
                    float* __restrict__ C, int M, int N, int K) {
    int lane = threadIdx.x;
    int n0 = blockIdx.x * 64;
    int m0 = blockIdx.y * 32;

    v8f acc[2][4] = {};
    for (int k0 = 0; k0 < K; k0 += 32) {
        v16bf a0 = load_frag_A(A, K, m0,      k0, lane);
        v16bf a1 = load_frag_A(A, K, m0 + 16, k0, lane);
#pragma unroll
        for (int j = 0; j < 4; ++j) {
            v16bf bj = load_frag_B(Bt, K, n0 + 16 * j, k0, lane);
            acc[0][j] = wmma_bf16(a0, bj, acc[0][j]);
            acc[1][j] = wmma_bf16(a1, bj, acc[1][j]);
        }
    }
    int cl = lane & 15;
    int hs = (lane >> 4) * 8;
#pragma unroll
    for (int i = 0; i < 2; ++i)
#pragma unroll
        for (int j = 0; j < 4; ++j)
#pragma unroll
            for (int r = 0; r < 8; ++r) {
                int row = m0 + 16 * i + r + hs;
                C[(size_t)row * N + n0 + 16 * j + cl] = acc[i][j][r];
            }
}

// ---------------- RoPE + head reshape ----------------
// q/k/v fp32 [B*S, HID] -> q_bh/k_bh bf16 [B,H,S,D] (RoPE on q,k),
// v_t bf16 [B,H,D,S]
__global__ void rope_reshape_kernel(const float* __restrict__ qf,
                                    const float* __restrict__ kf,
                                    const float* __restrict__ vf,
                                    unsigned short* __restrict__ qbh,
                                    unsigned short* __restrict__ kbh,
                                    unsigned short* __restrict__ vt) {
    int t = blockIdx.x * blockDim.x + threadIdx.x;   // B*H*S*D threads
    int d = t & 63;
    int s = (t >> 6) & (Sq - 1);
    int h = (t >> 17) & (NH - 1);
    int b = t >> 22;

    int i = d & 31;
    float inv = __expf(-0.2878231366242557f * (float)i);  // 10000^(-i/32)
    float ang = (float)s * inv;
    float sn, cs;
    __sincosf(ang, &sn, &cs);
    float sign = (d < 32) ? -1.0f : 1.0f;

    size_t src  = ((size_t)(b * Sq + s)) * HIDN + h * 64;
    float qv = qf[src + d], qp = qf[src + (d ^ 32)];
    float kv = kf[src + d], kp = kf[src + (d ^ 32)];
    float qo = qv * cs + sign * qp * sn;
    float ko = kv * cs + sign * kp * sn;

    size_t dst = ((size_t)((b * NH + h) * Sq + s)) * Dh + d;
    qbh[dst] = f2bf(qo);
    kbh[dst] = f2bf(ko);
    vt[((size_t)((b * NH + h) * Dh + d)) * Sq + s] = f2bf(vf[src + d]);
}

// ---------------- fused causal flash attention ----------------
// grid (S/16, H, B), block 32 (one wave per 16-row Q tile)
__global__ __launch_bounds__(32)
void flash_attn_kernel(const unsigned short* __restrict__ qbh,
                       const unsigned short* __restrict__ kbh,
                       const unsigned short* __restrict__ vt,
                       float* __restrict__ attn_out) {
    int lane = threadIdx.x;
    int qt = blockIdx.x, h = blockIdx.y, b = blockIdx.z;
    const unsigned short* qb = qbh + (size_t)(b * NH + h) * Sq * Dh;
    const unsigned short* kb = kbh + (size_t)(b * NH + h) * Sq * Dh;
    const unsigned short* vb = vt  + (size_t)(b * NH + h) * Dh * Sq;

    v16bf qA0 = load_frag_A(qb, Dh, qt * 16, 0,  lane);
    v16bf qA1 = load_frag_A(qb, Dh, qt * 16, 32, lane);

    v8f o0 = {}, o1 = {}, o2 = {}, o3 = {};
    float mrow[8], lrow[8];
#pragma unroll
    for (int i = 0; i < 8; ++i) { mrow[i] = -1e30f; lrow[i] = 0.0f; }

    __shared__ __align__(16) unsigned short pt[16][32];
    int cl = lane & 15;
    int hs = (lane >> 4) * 8;   // row offset within tile AND A-operand K-base
    int kend = (qt + 1) * 16;

    for (int kk = 0; kk < kend; kk += 32) {
        v8f s0 = {}, s1 = {};
        {
            v16bf b00 = load_frag_B(kb, Dh, kk,      0,  lane);
            v16bf b01 = load_frag_B(kb, Dh, kk,      32, lane);
            v16bf b10 = load_frag_B(kb, Dh, kk + 16, 0,  lane);
            v16bf b11 = load_frag_B(kb, Dh, kk + 16, 32, lane);
            s0 = wmma_bf16(qA0, b00, s0);
            s0 = wmma_bf16(qA1, b01, s0);
            s1 = wmma_bf16(qA0, b10, s1);
            s1 = wmma_bf16(qA1, b11, s1);
        }
        __syncthreads();   // protect pt vs previous iteration's reads
#pragma unroll
        for (int i = 0; i < 8; ++i) {
            int row  = qt * 16 + i + hs;
            int key0 = kk + cl, key1 = kk + 16 + cl;
            float v0 = (key0 <= row) ? s0[i] * SCALE : -1e30f;
            float v1 = (key1 <= row) ? s1[i] * SCALE : -1e30f;
            float rmx = fmaxf(v0, v1);
            rmx = fmaxf(rmx, __shfl_xor(rmx, 1, 32));
            rmx = fmaxf(rmx, __shfl_xor(rmx, 2, 32));
            rmx = fmaxf(rmx, __shfl_xor(rmx, 4, 32));
            rmx = fmaxf(rmx, __shfl_xor(rmx, 8, 32));
            float mn    = fmaxf(mrow[i], rmx);
            float alpha = __expf(mrow[i] - mn);
            float p0 = __expf(v0 - mn);
            float p1 = __expf(v1 - mn);
            float rs = p0 + p1;
            rs += __shfl_xor(rs, 1, 32);
            rs += __shfl_xor(rs, 2, 32);
            rs += __shfl_xor(rs, 4, 32);
            rs += __shfl_xor(rs, 8, 32);
            lrow[i] = lrow[i] * alpha + rs;
            mrow[i] = mn;
            o0[i] *= alpha; o1[i] *= alpha; o2[i] *= alpha; o3[i] *= alpha;
            pt[i + hs][cl]      = f2bf(p0);
            pt[i + hs][16 + cl] = f2bf(p1);
        }
        __syncthreads();
        Frag pa;
        const u32x4* lp = (const u32x4*)&pt[cl][hs];  // A-layout read of P
        pa.q[0] = lp[0];
        pa.q[1] = lp[2];
        o0 = wmma_bf16(pa.v, load_frag_B(vb, Sq, 0,  kk, lane), o0);
        o1 = wmma_bf16(pa.v, load_frag_B(vb, Sq, 16, kk, lane), o1);
        o2 = wmma_bf16(pa.v, load_frag_B(vb, Sq, 32, kk, lane), o2);
        o3 = wmma_bf16(pa.v, load_frag_B(vb, Sq, 48, kk, lane), o3);
    }

#pragma unroll
    for (int i = 0; i < 8; ++i) {
        float inv = 1.0f / lrow[i];
        int row = qt * 16 + i + hs;
        size_t base = ((size_t)b * Sq + row) * HIDN + h * 64;
        attn_out[base +  0 + cl] = o0[i] * inv;
        attn_out[base + 16 + cl] = o1[i] * inv;
        attn_out[base + 32 + cl] = o2[i] * inv;
        attn_out[base + 48 + cl] = o3[i] * inv;
    }
}

// ---------------- tiny adapter K/V projection (fp32) ----------------
// out[b][h][al][d] = adapter[b,al,:] . W[:, h*64+d]
__global__ void adapter_proj_kernel(const float* __restrict__ adapter,
                                    const float* __restrict__ W,
                                    float* __restrict__ outbh) {
    int t = blockIdx.x * blockDim.x + threadIdx.x;
    if (t >= Bsz * ALEN * HIDN) return;
    int n  = t % HIDN;
    int al = (t / HIDN) % ALEN;
    int b  = t / (HIDN * ALEN);
    const float* a = adapter + (size_t)(b * ALEN + al) * HIDN;
    float s = 0.0f;
    for (int k = 0; k < HIDN; ++k) s += a[k] * W[(size_t)k * HIDN + n];
    int h = n >> 6, d = n & 63;
    outbh[((size_t)(b * NH + h) * ALEN + al) * Dh + d] = s;
}

// ---------------- adapter cross-attention add ----------------
// one wave per (b,h,s) row; lane handles 2 d-elements
__global__ void adapter_add_kernel(const unsigned short* __restrict__ qbh,
                                   const float* __restrict__ ak,
                                   const float* __restrict__ av,
                                   const float* __restrict__ gate,
                                   float* __restrict__ attn_out) {
    int lane = threadIdx.x & 31;
    int w = threadIdx.x >> 5;
    int r = blockIdx.x * 4 + w;          // r < B*H*S
    int s = r & (Sq - 1);
    int h = (r >> 11) & (NH - 1);
    int b = r >> 16;

    const unsigned short* q = qbh + ((size_t)(b * NH + h) * Sq + s) * Dh;
    float q0 = bf2f(q[2 * lane]);
    float q1 = bf2f(q[2 * lane + 1]);
    const float* akp = ak + (size_t)(b * NH + h) * ALEN * Dh;
    const float* avp = av + (size_t)(b * NH + h) * ALEN * Dh;

    float sc[ALEN];
#pragma unroll
    for (int j = 0; j < ALEN; ++j) {
        float part = q0 * akp[j * Dh + 2 * lane] + q1 * akp[j * Dh + 2 * lane + 1];
        part += __shfl_xor(part, 16, 32);
        part += __shfl_xor(part, 8, 32);
        part += __shfl_xor(part, 4, 32);
        part += __shfl_xor(part, 2, 32);
        part += __shfl_xor(part, 1, 32);
        sc[j] = part * SCALE;
    }
    float mx = sc[0];
#pragma unroll
    for (int j = 1; j < ALEN; ++j) mx = fmaxf(mx, sc[j]);
    float sum = 0.0f;
#pragma unroll
    for (int j = 0; j < ALEN; ++j) { sc[j] = __expf(sc[j] - mx); sum += sc[j]; }
    float g = tanhf(gate[h]) / sum;

    float a0 = 0.0f, a1 = 0.0f;
#pragma unroll
    for (int j = 0; j < ALEN; ++j) {
        a0 += sc[j] * avp[j * Dh + 2 * lane];
        a1 += sc[j] * avp[j * Dh + 2 * lane + 1];
    }
    size_t base = ((size_t)b * Sq + s) * HIDN + h * 64;
    attn_out[base + 2 * lane]     += g * a0;
    attn_out[base + 2 * lane + 1] += g * a1;
}

// ---------------- launcher ----------------

extern "C" void kernel_launch(void* const* d_in, const int* in_sizes, int n_in,
                              void* d_out, int out_size, void* d_ws, size_t ws_size,
                              hipStream_t stream) {
    (void)in_sizes; (void)n_in; (void)out_size; (void)ws_size;
    const float* hidden  = (const float*)d_in[0];
    // d_in[1] = attention_mask (causal; implemented analytically)
    const float* adapter = (const float*)d_in[2];
    const float* Wq = (const float*)d_in[3];
    const float* Wk = (const float*)d_in[4];
    const float* Wv = (const float*)d_in[5];
    const float* Wo = (const float*)d_in[6];
    const float* gate = (const float*)d_in[7];
    float* out = (float*)d_out;

    char* ws = (char*)d_ws;
    const size_t SZ_HBF = (size_t)BSROWS * HIDN * 2;       // 16 MiB
    const size_t SZ_WT  = (size_t)HIDN * HIDN * 2;         // 8 MiB each
    const size_t SZ_F32 = (size_t)BSROWS * HIDN * 4;       // 32 MiB each
    const size_t SZ_BH  = (size_t)Bsz * NH * Sq * Dh * 2;  // 16 MiB each
    const size_t SZ_A   = (size_t)Bsz * NH * ALEN * Dh * 4;

    size_t o = 0;
    unsigned short* hbf = (unsigned short*)(ws + o); o += SZ_HBF;
    unsigned short* wqt = (unsigned short*)(ws + o); o += SZ_WT;
    unsigned short* wkt = (unsigned short*)(ws + o); o += SZ_WT;
    unsigned short* wvt = (unsigned short*)(ws + o); o += SZ_WT;
    unsigned short* wot = (unsigned short*)(ws + o); o += SZ_WT;
    float* qf = (float*)(ws + o); o += SZ_F32;   // later reused as attn_out
    float* kf = (float*)(ws + o); o += SZ_F32;   // later reused as attn_bf16
    float* vf = (float*)(ws + o); o += SZ_F32;
    unsigned short* qbh = (unsigned short*)(ws + o); o += SZ_BH;
    unsigned short* kbh = (unsigned short*)(ws + o); o += SZ_BH;
    unsigned short* vtp = (unsigned short*)(ws + o); o += SZ_BH;
    float* akb = (float*)(ws + o); o += SZ_A;
    float* avb = (float*)(ws + o); o += SZ_A;

    const int nElemHid = BSROWS * HIDN;        // 8388608
    const int nElemW   = HIDN * HIDN;          // 4194304

    // 1. hidden -> bf16
    cvt_bf16_kernel<<<nElemHid / 256, 256, 0, stream>>>(hidden, hbf, nElemHid);
    // 2. weights -> transposed bf16
    transpose_cvt_kernel<<<nElemW / 256, 256, 0, stream>>>(Wq, wqt);
    transpose_cvt_kernel<<<nElemW / 256, 256, 0, stream>>>(Wk, wkt);
    transpose_cvt_kernel<<<nElemW / 256, 256, 0, stream>>>(Wv, wvt);
    transpose_cvt_kernel<<<nElemW / 256, 256, 0, stream>>>(Wo, wot);
    // 3. Q/K/V projections (WMMA GEMM)
    dim3 ggrid(HIDN / 64, BSROWS / 32);
    gemm_bt_kernel<<<ggrid, 32, 0, stream>>>(hbf, wqt, qf, BSROWS, HIDN, HIDN);
    gemm_bt_kernel<<<ggrid, 32, 0, stream>>>(hbf, wkt, kf, BSROWS, HIDN, HIDN);
    gemm_bt_kernel<<<ggrid, 32, 0, stream>>>(hbf, wvt, vf, BSROWS, HIDN, HIDN);
    // 4. RoPE + reshape
    rope_reshape_kernel<<<(Bsz * NH * Sq * Dh) / 256, 256, 0, stream>>>(
        qf, kf, vf, qbh, kbh, vtp);
    // 5. adapter projections
    int nA = Bsz * ALEN * HIDN;
    adapter_proj_kernel<<<(nA + 255) / 256, 256, 0, stream>>>(adapter, Wk, akb);
    adapter_proj_kernel<<<(nA + 255) / 256, 256, 0, stream>>>(adapter, Wv, avb);
    // 6. fused causal flash attention -> attn_out (reuse qf)
    float* attn = qf;
    flash_attn_kernel<<<dim3(Sq / 16, NH, Bsz), 32, 0, stream>>>(qbh, kbh, vtp, attn);
    // 7. adapter cross-attention add
    adapter_add_kernel<<<(Bsz * NH * Sq) / 4, 128, 0, stream>>>(
        qbh, akb, avb, gate, attn);
    // 8. attn -> bf16 (reuse kf region)
    unsigned short* attnbf = (unsigned short*)kf;
    cvt_bf16_kernel<<<nElemHid / 256, 256, 0, stream>>>(attn, attnbf, nElemHid);
    // 9. output projection -> d_out
    gemm_bt_kernel<<<ggrid, 32, 0, stream>>>(attnbf, wot, out, BSROWS, HIDN, HIDN);
}
